// HybridCameraBEVModel_26130581028922
// MI455X (gfx1250) — compile-verified, matrix operand order checked by reading, not run on hardware
//
#include <hip/hip_runtime.h>
#include <hip/hip_bf16.h>
#include <cstdint>
#include <cstddef>

// ---------------------------------------------------------------------------
// Multi-scale deformable attention for MI455X (gfx1250, wave32, WMMA).
//  - All 4 GEMMs run through v_wmma_f32_16x16x32_bf16 (bf16 in, f32 accum),
//    16x64 output per wave (4 accumulators reuse one A fragment -> 4 wmma
//    per 10 b128 loads instead of 1 per 4).
//  - Bilinear-gather core is one wave per (batch,query,head), lane = channel;
//    divisions algebraically folded out of the sampling-location math.
// ---------------------------------------------------------------------------

constexpr int BS  = 4;
constexpr int NQ  = 10000;
constexpr int EMB = 256;
constexpr int NH  = 8;
constexpr int HD  = 32;
constexpr int NL  = 4;
constexpr int NP  = 8;
constexpr int NV  = 30825;   // 116*200 + 58*100 + 29*50 + 15*25

// WMMA fragment types (probe-confirmed signatures)
typedef __bf16 v16bf __attribute__((ext_vector_type(16)));
typedef float  v8f   __attribute__((ext_vector_type(8)));
typedef unsigned int u32x4 __attribute__((ext_vector_type(4)));

union FragBF { v16bf v; u32x4 q[2]; };
union FragF  { v8f   v; float f[8]; };

__device__ __forceinline__ unsigned short f32_to_bf16_bits(float f) {
  uint32_t x = __float_as_uint(f);
  uint32_t r = x + 0x7FFFu + ((x >> 16) & 1u);   // round-to-nearest-even
  return (unsigned short)(r >> 16);
}

// ---------------------------------------------------------------------------
// f32 -> bf16 bulk conversion (grid-stride)
// ---------------------------------------------------------------------------
__global__ void cvt_bf16_kernel(const float* __restrict__ src,
                                unsigned short* __restrict__ dst, size_t n) {
  size_t i = (size_t)blockIdx.x * blockDim.x + threadIdx.x;
  size_t stride = (size_t)gridDim.x * blockDim.x;
  for (; i < n; i += stride) dst[i] = f32_to_bf16_bits(src[i]);
}

// ---------------------------------------------------------------------------
// Weight pack: W (K x N, f32 row-major) -> Wp (N x K, bf16) i.e. W^T
// so B fragments are contiguous-in-K loads per lane.
// ---------------------------------------------------------------------------
__global__ void pack_weight_kernel(const float* __restrict__ W,
                                   unsigned short* __restrict__ Wp,
                                   int K, int N) {
  size_t total = (size_t)K * N;
  size_t i = (size_t)blockIdx.x * blockDim.x + threadIdx.x;
  size_t stride = (size_t)gridDim.x * blockDim.x;
  for (; i < total; i += stride) {
    int n = (int)(i / K);
    int k = (int)(i % K);
    Wp[i] = f32_to_bf16_bits(W[(size_t)k * N + n]);
  }
}

// ---------------------------------------------------------------------------
// WMMA GEMM:  C[M,N] = A[M,K](bf16) * Wp[N,K](bf16)^T + bias[N] (+ residual)
// One 16x64 output strip per wave (4 x 16x16 WMMA tiles sharing the A frag).
// K multiple of 32; N multiple of 64.
// Fragment layout per CDNA5 ISA 7.12.2 (16-bit A 16x32):
//   lanes 0-15 : row M=lane, K = kk+{0..7} (VGPR0-3), kk+{16..23} (VGPR4-7)
//   lanes 16-31: row M=lane-16, K offsets +8.
// B uses the symmetric layout with N in lanes (weights pre-transposed).
// ---------------------------------------------------------------------------
__global__ void wmma_gemm_bf16(const unsigned short* __restrict__ A,
                               const unsigned short* __restrict__ Bp,
                               const float* __restrict__ bias,
                               const float* __restrict__ residual,
                               float* __restrict__ C,
                               int M, int N, int K) {
  const int NG = N >> 6;                 // 64-wide N groups (4 WMMA tiles)
  const int Mt = (M + 15) >> 4;
  const int wave = blockIdx.x * (blockDim.x >> 5) + (threadIdx.x >> 5);
  if (wave >= Mt * NG) return;
  const int lane = threadIdx.x & 31;
  const int mt = wave / NG;
  const int ng = wave % NG;
  const int r16 = lane & 15;
  const int hi  = lane >> 4;             // selects +8 K offset per ISA layout

  int m_load = mt * 16 + r16;
  if (m_load >= M) m_load = M - 1;       // clamp loads; stores predicated below

  const unsigned short* arow = A + (size_t)m_load * K;
  const unsigned short* brow[4];
#pragma unroll
  for (int j = 0; j < 4; ++j)
    brow[j] = Bp + (size_t)(ng * 64 + j * 16 + r16) * K;

  FragF c[4];
#pragma unroll
  for (int j = 0; j < 4; ++j) {
    const float binit = bias ? bias[ng * 64 + j * 16 + r16] : 0.0f;
#pragma unroll
    for (int v = 0; v < 8; ++v) c[j].f[v] = binit;
  }

  for (int kk = 0; kk < K; kk += 32) {
    FragBF a;
    a.q[0] = *(const u32x4*)(arow + kk +      hi * 8);   // K = kk+hi8+{0..7}
    a.q[1] = *(const u32x4*)(arow + kk + 16 + hi * 8);   // K = kk+16+hi8+{0..7}
    if (kk + 32 < K) __builtin_prefetch(arow + kk + 32, 0, 1);  // global_prefetch_b8
#pragma unroll
    for (int j = 0; j < 4; ++j) {
      FragBF b;
      b.q[0] = *(const u32x4*)(brow[j] + kk +      hi * 8);
      b.q[1] = *(const u32x4*)(brow[j] + kk + 16 + hi * 8);
      c[j].v = __builtin_amdgcn_wmma_f32_16x16x32_bf16(
          /*neg_a=*/false, a.v, /*neg_b=*/false, b.v,
          /*c_mod=*/(short)0, c[j].v, /*reuse_a=*/false, /*reuse_b=*/false);
    }
  }

  // C/D layout: VGPR v, lane -> (M = v + 8*hi, N = lane&15) within tile
#pragma unroll
  for (int j = 0; j < 4; ++j) {
    const int nn = ng * 64 + j * 16 + r16;
#pragma unroll
    for (int v = 0; v < 8; ++v) {
      const int mm = mt * 16 + hi * 8 + v;
      if (mm < M) {
        float outv = c[j].f[v];
        if (residual) outv += residual[(size_t)mm * N + nn];
        C[(size_t)mm * N + nn] = outv;
      }
    }
  }
}

// ---------------------------------------------------------------------------
// Softmax over 32 logits per (b,q,head): one wave32 per row.
// ---------------------------------------------------------------------------
__global__ void softmax32_kernel(float* __restrict__ ATT, size_t rows) {
  const size_t row = ((size_t)blockIdx.x * blockDim.x + threadIdx.x) >> 5;
  if (row >= rows) return;
  const int lane = threadIdx.x & 31;
  float v = ATT[row * 32 + lane];
  float m = v;
#pragma unroll
  for (int s = 16; s > 0; s >>= 1) m = fmaxf(m, __shfl_xor(m, s, 32));
  float e = __expf(v - m);
  float sum = e;
#pragma unroll
  for (int s = 16; s > 0; s >>= 1) sum += __shfl_xor(sum, s, 32);
  ATT[row * 32 + lane] = e / sum;
}

// ---------------------------------------------------------------------------
// Deformable-attention gather core.
// grid = BS*NQ blocks, 256 threads = 8 waves; wave h handles head h, lane = d.
// V layout: (b, pos, h, d) f32 -> per-corner fetch is 128B coalesced per wave.
// Sampling-location math folded: (rx + ox/W)*W - 0.5 == rx*W + ox - 0.5,
// removing all divides/rcp from the hot loop.
// ---------------------------------------------------------------------------
__global__ void deform_attn_kernel(const float* __restrict__ V,
                                   const float* __restrict__ refp,  // (BS,NQ,NL,2)
                                   const float* __restrict__ OFF,   // (BS,NQ,512)
                                   const float* __restrict__ ATT,   // (BS,NQ,256)
                                   float* __restrict__ AOUT) {      // (BS,NQ,256)
  const int bq = blockIdx.x;
  const int b  = bq / NQ;
  const int h  = threadIdx.x >> 5;
  const int d  = threadIdx.x & 31;

  const float* off = OFF + (size_t)bq * (NH * NL * NP * 2) + h * (NL * NP * 2);
  const float* att = ATT + (size_t)bq * (NH * NL * NP) + h * (NL * NP);
  const float* ref = refp + (size_t)bq * (NL * 2);

  const int lvlH[NL] = {116, 58, 29, 15};
  const int lvlW[NL] = {200, 100, 50, 25};

  float acc = 0.0f;
  int start = 0;
#pragma unroll
  for (int l = 0; l < NL; ++l) {
    const int H = lvlH[l], W = lvlW[l];
    const float rxW = ref[l * 2 + 0] * (float)W - 0.5f;
    const float ryH = ref[l * 2 + 1] * (float)H - 0.5f;
    const float* vbase = V + (((size_t)b * NV + start) * NH + h) * HD + d;
#pragma unroll
    for (int p = 0; p < NP; ++p) {
      const float ox = off[(l * NP + p) * 2 + 0];
      const float oy = off[(l * NP + p) * 2 + 1];
      const float w  = att[l * NP + p];
      const float x = rxW + ox;          // == (rx + ox/W)*W - 0.5
      const float y = ryH + oy;
      const float xf = floorf(x), yf = floorf(y);
      const float wx = x - xf, wy = y - yf;
      const int x0 = (int)xf, y0 = (int)yf;

      auto fetch = [&](int xi, int yi) -> float {
        if (xi < 0 || xi >= W || yi < 0 || yi >= H) return 0.0f;
        return vbase[(size_t)(yi * W + xi) * (NH * HD)];
      };
      const float v00 = fetch(x0,     y0);
      const float v01 = fetch(x0 + 1, y0);
      const float v10 = fetch(x0,     y0 + 1);
      const float v11 = fetch(x0 + 1, y0 + 1);
      const float s = v00 * (1.0f - wx) * (1.0f - wy) + v01 * wx * (1.0f - wy) +
                      v10 * (1.0f - wx) * wy          + v11 * wx * wy;
      acc += w * s;
    }
    start += H * W;
  }
  AOUT[(size_t)bq * EMB + h * HD + d] = acc;
}

// ---------------------------------------------------------------------------
// Launch orchestration
// ---------------------------------------------------------------------------
static inline int gemm_blocks(int M, int N) {
  const long tiles = (long)((M + 15) / 16) * (N / 64);  // 16x64 per wave
  return (int)((tiles + 7) / 8);                        // 8 waves per block
}

extern "C" void kernel_launch(void* const* d_in, const int* in_sizes, int n_in,
                              void* d_out, int out_size, void* d_ws, size_t ws_size,
                              hipStream_t stream) {
  (void)in_sizes; (void)n_in; (void)out_size; (void)ws_size;

  const float* query = (const float*)d_in[0];   // (BS,NQ,EMB)
  const float* value = (const float*)d_in[1];   // (BS,NV,EMB)
  const float* refp  = (const float*)d_in[2];   // (BS,NQ,NL,2)
  // d_in[3] = spatial_shapes (compile-time constants here)
  const float* Wv    = (const float*)d_in[4];
  const float* bv    = (const float*)d_in[5];
  const float* Wo    = (const float*)d_in[6];
  const float* bo    = (const float*)d_in[7];
  const float* Wa    = (const float*)d_in[8];
  const float* ba    = (const float*)d_in[9];
  const float* Wout  = (const float*)d_in[10];
  const float* bout  = (const float*)d_in[11];
  float* out = (float*)d_out;

  // Workspace carve-up (all region sizes are multiples of 256 bytes)
  char* ws = (char*)d_ws;
  size_t o = 0;
  unsigned short* q_bf    = (unsigned short*)(ws + o); o += (size_t)BS * NQ * EMB * 2;
  unsigned short* v_bf    = (unsigned short*)(ws + o); o += (size_t)BS * NV * EMB * 2;
  unsigned short* Wv_p    = (unsigned short*)(ws + o); o += (size_t)EMB * EMB * 2;
  unsigned short* Wo_p    = (unsigned short*)(ws + o); o += (size_t)EMB * 512 * 2;
  unsigned short* Wa_p    = (unsigned short*)(ws + o); o += (size_t)EMB * EMB * 2;
  unsigned short* Wout_p  = (unsigned short*)(ws + o); o += (size_t)EMB * EMB * 2;
  float*          VPROJ   = (float*)(ws + o);          o += (size_t)BS * NV * EMB * 4;
  float*          OFFbuf  = (float*)(ws + o);          o += (size_t)BS * NQ * 512 * 4;
  float*          ATTbuf  = (float*)(ws + o);          o += (size_t)BS * NQ * EMB * 4;
  float*          AOUT    = (float*)(ws + o);          o += (size_t)BS * NQ * EMB * 4;
  unsigned short* aout_bf = (unsigned short*)(ws + o); o += (size_t)BS * NQ * EMB * 2;

  const int T = 256;

  // 1) Convert activations + pack weights to bf16
  cvt_bf16_kernel<<<4096, T, 0, stream>>>(query, q_bf, (size_t)BS * NQ * EMB);
  cvt_bf16_kernel<<<8192, T, 0, stream>>>(value, v_bf, (size_t)BS * NV * EMB);
  pack_weight_kernel<<<512, T, 0, stream>>>(Wv,   Wv_p,   EMB, EMB);
  pack_weight_kernel<<<1024, T, 0, stream>>>(Wo,  Wo_p,   EMB, 512);
  pack_weight_kernel<<<512, T, 0, stream>>>(Wa,   Wa_p,   EMB, EMB);
  pack_weight_kernel<<<512, T, 0, stream>>>(Wout, Wout_p, EMB, EMB);

  // 2) Value projection: (BS*NV, 256) x (256, 256) + bv
  wmma_gemm_bf16<<<gemm_blocks(BS * NV, EMB), T, 0, stream>>>(
      v_bf, Wv_p, bv, nullptr, VPROJ, BS * NV, EMB, EMB);

  // 3) Sampling offsets: (BS*NQ, 256) x (256, 512) + bo
  wmma_gemm_bf16<<<gemm_blocks(BS * NQ, 512), T, 0, stream>>>(
      q_bf, Wo_p, bo, nullptr, OFFbuf, BS * NQ, 512, EMB);

  // 4) Attention logits: (BS*NQ, 256) x (256, 256) + ba, then softmax over 32
  wmma_gemm_bf16<<<gemm_blocks(BS * NQ, EMB), T, 0, stream>>>(
      q_bf, Wa_p, ba, nullptr, ATTbuf, BS * NQ, EMB, EMB);
  {
    const size_t rows = (size_t)BS * NQ * NH;           // 320000 wave-rows
    const int blocks = (int)((rows * 32 + T - 1) / T);
    softmax32_kernel<<<blocks, T, 0, stream>>>(ATTbuf, rows);
  }

  // 5) Deformable gather core
  deform_attn_kernel<<<BS * NQ, T, 0, stream>>>(VPROJ, refp, OFFbuf, ATTbuf, AOUT);

  // 6) Output projection + residual: query + AOUT @ Wout + bout
  cvt_bf16_kernel<<<4096, T, 0, stream>>>(AOUT, aout_bf, (size_t)BS * NQ * EMB);
  wmma_gemm_bf16<<<gemm_blocks(BS * NQ, EMB), T, 0, stream>>>(
      aout_bf, Wout_p, bout, /*residual=*/query, out, BS * NQ, EMB, EMB);
}
